// TopKSAE_55087250538647
// MI455X (gfx1250) — compile-verified, matrix operand order checked by reading, not run on hardware
//
#include <hip/hip_runtime.h>
#include <hip/hip_bf16.h>

typedef __bf16 v16bf __attribute__((ext_vector_type(16)));
typedef float  v8f   __attribute__((ext_vector_type(8)));
typedef unsigned int u32x4 __attribute__((ext_vector_type(4)));
typedef int          i32x4 __attribute__((ext_vector_type(4)));
typedef int          i32x8 __attribute__((ext_vector_type(8)));

#define D_MODEL 768
#define N_FEAT  12288
#define TOPK    64
#define BATCH   8192

#define BM 128
#define BN 128
#define BK 32
#define LDA 40   // padded bf16 row stride (80B, 16B aligned)

// ---------------------------------------------------------------- TDM helper
// Build a 2D-tile D# (groups 0/1, zeroed 2/3) and issue TENSOR_LOAD_TO_LDS.
// bf16 elements (data_size=2B); LDS padding: 64B of data then 16B pad -> 80B rows.
__device__ __forceinline__ void tdm_load_tile_bf16(unsigned lds_byte_addr,
                                                   unsigned long long gaddr,
                                                   int tensor_d0, int tensor_d1,
                                                   long long stride0,
                                                   int tile_d0, int tile_d1)
{
    u32x4 g0;
    g0[0] = 1u;                                                   // count=1, user mode
    g0[1] = lds_byte_addr;                                        // lds_addr
    g0[2] = (unsigned)(gaddr & 0xFFFFFFFFull);                    // global_addr lo
    g0[3] = (unsigned)((gaddr >> 32) & 0x01FFFFFFull)             // global_addr hi (57b)
          | (2u << 30);                                           // type=2 ("image")
    i32x8 g1;
    g1[0] = (1 << 16)                                             // data_size = 2B
          | (1 << 20)                                             // pad_enable
          | (3 << 22)                                             // pad_interval: 16 DWORDs
          | (3 << 25);                                            // pad_amount:   4 DWORDs
    g1[1] = (tensor_d0 & 0xFFFF) << 16;                           // tensor_dim0 lo16
    g1[2] = (int)(((unsigned)tensor_d0 >> 16)
          | (((unsigned)tensor_d1 & 0xFFFFu) << 16));             // dim0 hi / dim1 lo
    g1[3] = (int)(((unsigned)tensor_d1 >> 16)
          | ((unsigned)tile_d0 << 16));                           // dim1 hi / tile_dim0
    g1[4] = tile_d1 & 0xFFFF;                                     // tile_dim1 (tile_dim2=0)
    g1[5] = (int)(stride0 & 0xFFFFFFFFll);                        // dim0_stride lo32
    g1[6] = (int)((stride0 >> 32) & 0xFFFFll);                    // dim0_stride hi16
    g1[7] = 0;
    i32x4 gz4 = {0, 0, 0, 0};
    i32x8 gz8 = {0, 0, 0, 0, 0, 0, 0, 0};
    // amdgpu-toolchain (clang-23 / therock-10.0) 6-arg form
    __builtin_amdgcn_tensor_load_to_lds(g0, g1, gz4, gz4, gz8, 0);
}

// -------- Kernel 0: f32 -> bf16 pre-conversion into workspace ---------------
__global__ __launch_bounds__(256)
void center_to_bf16(const float* __restrict__ x, const float* __restrict__ pre_bias,
                    __bf16* __restrict__ out)
{
    size_t i = ((size_t)blockIdx.x * 256 + threadIdx.x) * 8;
    int col = (int)(i % D_MODEL);                 // 768 % 8 == 0 -> stays in row
    float4 a = *(const float4*)(x + i);
    float4 b = *(const float4*)(x + i + 4);
    __bf16 t[8];
    t[0] = (__bf16)(a.x - pre_bias[col + 0]);
    t[1] = (__bf16)(a.y - pre_bias[col + 1]);
    t[2] = (__bf16)(a.z - pre_bias[col + 2]);
    t[3] = (__bf16)(a.w - pre_bias[col + 3]);
    t[4] = (__bf16)(b.x - pre_bias[col + 4]);
    t[5] = (__bf16)(b.y - pre_bias[col + 5]);
    t[6] = (__bf16)(b.z - pre_bias[col + 6]);
    t[7] = (__bf16)(b.w - pre_bias[col + 7]);
    *(uint4*)(out + i) = *(const uint4*)t;
}

__global__ __launch_bounds__(256)
void to_bf16(const float* __restrict__ in, __bf16* __restrict__ out)
{
    size_t i = ((size_t)blockIdx.x * 256 + threadIdx.x) * 8;
    float4 a = *(const float4*)(in + i);
    float4 b = *(const float4*)(in + i + 4);
    __bf16 t[8];
    t[0] = (__bf16)a.x; t[1] = (__bf16)a.y; t[2] = (__bf16)a.z; t[3] = (__bf16)a.w;
    t[4] = (__bf16)b.x; t[5] = (__bf16)b.y; t[6] = (__bf16)b.z; t[7] = (__bf16)b.w;
    *(uint4*)(out + i) = *(const uint4*)t;
}

// -------- Kernel 1 (primary): TDM double-buffered bf16 WMMA GEMM ------------
// z = centered_bf16 @ W_enc_bf16^T + b_enc
__global__ __launch_bounds__(256)
void sae_encode_wmma_tdm(const __bf16* __restrict__ xbf,   // [B, 768]
                         const __bf16* __restrict__ wbf,   // [F, 768]
                         const float* __restrict__ b_enc,
                         float* __restrict__ z)
{
    __shared__ __align__(16) __bf16 ldsA[2][BM * LDA];
    __shared__ __align__(16) __bf16 ldsB[2][BN * LDA];

    const int tid  = threadIdx.x;
    const int lane = tid & 31;
    const int w    = tid >> 5;
    const int wm   = w & 1;
    const int wn   = w >> 1;
    const int blockN = blockIdx.x * BN;
    const int blockM = blockIdx.y * BM;
    const bool issuer = (tid < 32);           // wave 0 drives the TDM

    v8f acc[4][2];
    #pragma unroll
    for (int mt = 0; mt < 4; ++mt)
        #pragma unroll
        for (int nt = 0; nt < 2; ++nt)
            #pragma unroll
            for (int e = 0; e < 8; ++e) acc[mt][nt][e] = 0.0f;

    const int off = (lane >= 16) ? 8 : 0;     // ISA 16-bit A/B per-lane K halves
    const int rA  = lane & 15;

    if (issuer) {
        tdm_load_tile_bf16((unsigned)(size_t)&ldsA[0][0],
                           (unsigned long long)(size_t)(xbf + (size_t)blockM * D_MODEL),
                           D_MODEL, BATCH, D_MODEL, BK, BM);
        tdm_load_tile_bf16((unsigned)(size_t)&ldsB[0][0],
                           (unsigned long long)(size_t)(wbf + (size_t)blockN * D_MODEL),
                           D_MODEL, N_FEAT, D_MODEL, BK, BN);
    }

    const int NK = D_MODEL / BK;              // 24
    for (int it = 0; it < NK; ++it) {
        const int cur = it & 1;
        if (it + 1 < NK) {
            if (issuer) {
                const int k0 = (it + 1) * BK;
                tdm_load_tile_bf16((unsigned)(size_t)&ldsA[cur ^ 1][0],
                                   (unsigned long long)(size_t)(xbf + (size_t)blockM * D_MODEL + k0),
                                   D_MODEL, BATCH, D_MODEL, BK, BM);
                tdm_load_tile_bf16((unsigned)(size_t)&ldsB[cur ^ 1][0],
                                   (unsigned long long)(size_t)(wbf + (size_t)blockN * D_MODEL + k0),
                                   D_MODEL, N_FEAT, D_MODEL, BK, BN);
                __builtin_amdgcn_s_wait_tensorcnt(2);  // current pair done (in-order)
            }
        } else {
            if (issuer) __builtin_amdgcn_s_wait_tensorcnt(0);
        }
        __syncthreads();                      // publish current buffer

        v16bf afrag[4], bfrag[2];
        #pragma unroll
        for (int mt = 0; mt < 4; ++mt) {
            const __bf16* p = &ldsA[cur][(wm * 64 + mt * 16 + rA) * LDA];
            ((uint4*)&afrag[mt])[0] = *(const uint4*)(p + off);
            ((uint4*)&afrag[mt])[1] = *(const uint4*)(p + off + 16);
        }
        #pragma unroll
        for (int nt = 0; nt < 2; ++nt) {
            const __bf16* p = &ldsB[cur][(wn * 32 + nt * 16 + rA) * LDA];
            ((uint4*)&bfrag[nt])[0] = *(const uint4*)(p + off);
            ((uint4*)&bfrag[nt])[1] = *(const uint4*)(p + off + 16);
        }
        #pragma unroll
        for (int mt = 0; mt < 4; ++mt)
            #pragma unroll
            for (int nt = 0; nt < 2; ++nt)
                acc[mt][nt] = __builtin_amdgcn_wmma_f32_16x16x32_bf16(
                    false, afrag[mt], false, bfrag[nt],
                    (short)0, acc[mt][nt], false, false);
        __syncthreads();                      // readers done before next overwrite
    }

    const int cn   = lane & 15;
    const int rofs = (lane >= 16) ? 8 : 0;
    #pragma unroll
    for (int mt = 0; mt < 4; ++mt)
        #pragma unroll
        for (int nt = 0; nt < 2; ++nt) {
            int col  = blockN + wn * 32 + nt * 16 + cn;
            float bb = b_enc[col];
            #pragma unroll
            for (int r = 0; r < 8; ++r) {
                int row = blockM + wm * 64 + mt * 16 + rofs + r;
                z[(size_t)row * N_FEAT + col] = acc[mt][nt][r] + bb;
            }
        }
}

// -------- Kernel 1 (fallback, no workspace): fused-convert WMMA GEMM --------
__global__ __launch_bounds__(256)
void sae_encode_wmma(const float* __restrict__ x,
                     const float* __restrict__ pre_bias,
                     const float* __restrict__ W_enc,
                     const float* __restrict__ b_enc,
                     float* __restrict__ z)
{
    __shared__ __align__(16) __bf16 ldsA[BM * LDA];
    __shared__ __align__(16) __bf16 ldsB[BN * LDA];

    const int tid  = threadIdx.x;
    const int lane = tid & 31;
    const int w    = tid >> 5;
    const int wm   = w & 1;
    const int wn   = w >> 1;
    const int blockN = blockIdx.x * BN;
    const int blockM = blockIdx.y * BM;

    v8f acc[4][2];
    #pragma unroll
    for (int mt = 0; mt < 4; ++mt)
        #pragma unroll
        for (int nt = 0; nt < 2; ++nt)
            #pragma unroll
            for (int e = 0; e < 8; ++e) acc[mt][nt][e] = 0.0f;

    const int off = (lane >= 16) ? 8 : 0;
    const int rA  = lane & 15;

    for (int k0 = 0; k0 < D_MODEL; k0 += BK) {
        #pragma unroll
        for (int i = 0; i < 4; ++i) {
            int v   = tid + i * 256;
            int row = v >> 3;
            int c4  = (v & 7) << 2;
            const float4 xa = *(const float4*)(x + (size_t)(blockM + row) * D_MODEL + k0 + c4);
            __bf16* da = ldsA + row * LDA + c4;
            da[0] = (__bf16)(xa.x - pre_bias[k0 + c4 + 0]);
            da[1] = (__bf16)(xa.y - pre_bias[k0 + c4 + 1]);
            da[2] = (__bf16)(xa.z - pre_bias[k0 + c4 + 2]);
            da[3] = (__bf16)(xa.w - pre_bias[k0 + c4 + 3]);
            const float4 wb = *(const float4*)(W_enc + (size_t)(blockN + row) * D_MODEL + k0 + c4);
            __bf16* db = ldsB + row * LDA + c4;
            db[0] = (__bf16)wb.x; db[1] = (__bf16)wb.y;
            db[2] = (__bf16)wb.z; db[3] = (__bf16)wb.w;
        }
        __syncthreads();

        v16bf afrag[4], bfrag[2];
        #pragma unroll
        for (int mt = 0; mt < 4; ++mt) {
            const __bf16* p = ldsA + (wm * 64 + mt * 16 + rA) * LDA;
            ((uint4*)&afrag[mt])[0] = *(const uint4*)(p + off);
            ((uint4*)&afrag[mt])[1] = *(const uint4*)(p + off + 16);
        }
        #pragma unroll
        for (int nt = 0; nt < 2; ++nt) {
            const __bf16* p = ldsB + (wn * 32 + nt * 16 + rA) * LDA;
            ((uint4*)&bfrag[nt])[0] = *(const uint4*)(p + off);
            ((uint4*)&bfrag[nt])[1] = *(const uint4*)(p + off + 16);
        }
        #pragma unroll
        for (int mt = 0; mt < 4; ++mt)
            #pragma unroll
            for (int nt = 0; nt < 2; ++nt)
                acc[mt][nt] = __builtin_amdgcn_wmma_f32_16x16x32_bf16(
                    false, afrag[mt], false, bfrag[nt],
                    (short)0, acc[mt][nt], false, false);
        __syncthreads();
    }

    const int cn   = lane & 15;
    const int rofs = (lane >= 16) ? 8 : 0;
    #pragma unroll
    for (int mt = 0; mt < 4; ++mt)
        #pragma unroll
        for (int nt = 0; nt < 2; ++nt) {
            int col  = blockN + wn * 32 + nt * 16 + cn;
            float bb = b_enc[col];
            #pragma unroll
            for (int r = 0; r < 8; ++r) {
                int row = blockM + wm * 64 + mt * 16 + rofs + r;
                z[(size_t)row * N_FEAT + col] = acc[mt][nt][r] + bb;
            }
        }
}

// -------- Kernel 2: exact top-64 radix select + scatter + fused sparse decode
__device__ __forceinline__ unsigned fkey(float x) {
    unsigned u = __float_as_uint(x);
    return u ^ ((unsigned)(((int)u) >> 31) | 0x80000000u);  // monotonic key
}

__global__ __launch_bounds__(256)
void sae_topk_decode(float* __restrict__ sparse,          // in: z rows, out: sparse rows
                     const float* __restrict__ W_dec,     // [768, 12288]
                     const float* __restrict__ pre_bias,  // [768]
                     float* __restrict__ xhat)            // [B, 768]
{
    __shared__ float    rowv[N_FEAT];
    __shared__ unsigned hist[256];
    __shared__ unsigned s_prefix, s_remaining;
    __shared__ int      s_tie_quota, s_ties_taken, s_nsel;
    __shared__ int      lidx[TOPK];
    __shared__ float    lval[TOPK];

    const int b   = blockIdx.x;
    const int tid = threadIdx.x;
    float* zrow = sparse + (size_t)b * N_FEAT;

    for (int i = tid; i < N_FEAT; i += 256) rowv[i] = zrow[i];
    if (tid == 0) { s_prefix = 0u; s_remaining = TOPK; }
    __syncthreads();

    unsigned mask = 0u;
    for (int shift = 24; shift >= 0; shift -= 8) {
        hist[tid] = 0u;
        __syncthreads();
        const unsigned prefix = s_prefix;
        for (int i = tid; i < N_FEAT; i += 256) {
            unsigned u = fkey(rowv[i]);
            if ((u & mask) == prefix)
                atomicAdd(&hist[(u >> shift) & 0xFFu], 1u);
        }
        __syncthreads();
        if (tid == 0) {
            unsigned rem = s_remaining, cum = 0u;
            int bkt = 255;
            for (; bkt > 0; --bkt) {
                cum += hist[bkt];
                if (cum >= rem) break;
            }
            if (cum < rem) cum += hist[0];
            s_remaining = rem - (cum - hist[bkt]);
            s_prefix    = prefix | ((unsigned)bkt << shift);
        }
        mask |= (0xFFu << shift);
        __syncthreads();
    }

    const unsigned thkey = s_prefix;
    if (tid == 0) { s_tie_quota = (int)s_remaining; s_ties_taken = 0; s_nsel = 0; }
    __syncthreads();

    for (int i = tid; i < N_FEAT; i += 256) {
        float v = rowv[i];
        unsigned u = fkey(v);
        bool keep = (u > thkey);
        if (!keep && u == thkey)
            keep = (atomicAdd(&s_ties_taken, 1) < s_tie_quota);
        zrow[i] = keep ? fmaxf(v, 0.0f) : 0.0f;
        if (keep && v > 0.0f) {
            int p = atomicAdd(&s_nsel, 1);
            if (p < TOPK) { lidx[p] = i; lval[p] = v; }
        }
    }
    __syncthreads();

    const int nsel = (s_nsel > TOPK) ? TOPK : s_nsel;
    for (int d = tid; d < D_MODEL; d += 256) {
        float accv = pre_bias[d];
        const float* wrow = W_dec + (size_t)d * N_FEAT;
        for (int j = 0; j < nsel; ++j)
            accv += lval[j] * wrow[lidx[j]];
        xhat[(size_t)b * D_MODEL + d] = accv;
    }
}

extern "C" void kernel_launch(void* const* d_in, const int* in_sizes, int n_in,
                              void* d_out, int out_size, void* d_ws, size_t ws_size,
                              hipStream_t stream) {
    const float* x        = (const float*)d_in[0];
    const float* pre_bias = (const float*)d_in[1];
    const float* W_enc    = (const float*)d_in[2];
    const float* b_enc    = (const float*)d_in[3];
    const float* W_dec    = (const float*)d_in[4];

    float* out    = (float*)d_out;
    float* xhat   = out;                              // [B, 768]
    float* sparse = out + (size_t)BATCH * D_MODEL;    // [B, 12288]

    const size_t xbf_elems = (size_t)BATCH * D_MODEL;
    const size_t wbf_elems = (size_t)N_FEAT * D_MODEL;
    const size_t need = (xbf_elems + wbf_elems) * sizeof(__bf16);

    dim3 g1(N_FEAT / BN, BATCH / BM);                 // 96 x 64 tiles

    if (ws_size >= need) {
        __bf16* xbf = (__bf16*)d_ws;
        __bf16* wbf = xbf + xbf_elems;
        center_to_bf16<<<(unsigned)(xbf_elems / 8 / 256), 256, 0, stream>>>(x, pre_bias, xbf);
        to_bf16<<<(unsigned)(wbf_elems / 8 / 256), 256, 0, stream>>>(W_enc, wbf);
        sae_encode_wmma_tdm<<<g1, 256, 0, stream>>>(xbf, wbf, b_enc, sparse);
    } else {
        sae_encode_wmma<<<g1, 256, 0, stream>>>(x, pre_bias, W_enc, b_enc, sparse);
    }
    sae_topk_decode<<<BATCH, 256, 0, stream>>>(sparse, W_dec, pre_bias, xhat);
}
